// CausalSelfAttention_4037269258312
// MI455X (gfx1250) — matrix operand order repaired and checked by reading, not compiled, not run on hardware
//
#include <hip/hip_runtime.h>
#include <hip/hip_bf16.h>
#include <math.h>

typedef _Float16 v8h  __attribute__((ext_vector_type(8)));
typedef _Float16 v16h __attribute__((ext_vector_type(16)));
typedef float    v8f  __attribute__((ext_vector_type(8)));

#define WMMA_F16(a, b, c) \
  __builtin_amdgcn_wmma_f32_16x16x32_f16(false, (a), false, (b), (short)0, (c), false, false)

// Problem constants
constexpr int Bq = 8, Sq = 1024, Dq = 768, Hq = 12, HSq = 64, N3 = 3 * Dq; // 2304
constexpr int Mrows = Bq * Sq;                                            // 8192

// ---------------------------------------------------------------------------
// fp32 -> fp16 convert
__global__ __launch_bounds__(256) void cvt_f32_f16(const float* __restrict__ in,
                                                   _Float16* __restrict__ out, int n) {
  int i = blockIdx.x * blockDim.x + threadIdx.x;
  if (i < n) out[i] = (_Float16)in[i];
}

// W[K][N] (fp32) -> Wt[N][K] (fp16), LDS-tiled so both sides stay coalesced
__global__ __launch_bounds__(256) void cvt_transpose(const float* __restrict__ W,
                                                     _Float16* __restrict__ Wt,
                                                     int K, int N) {
  __shared__ _Float16 tile[32][33];
  const int tx = threadIdx.x, ty = threadIdx.y;      // 32 x 8
  const int k0 = blockIdx.x * 32, n0 = blockIdx.y * 32;
#pragma unroll
  for (int j = 0; j < 4; ++j) {
    const int k = k0 + ty + 8 * j, n = n0 + tx;
    if (k < K && n < N) tile[ty + 8 * j][tx] = (_Float16)W[(size_t)k * N + n];
  }
  __syncthreads();
#pragma unroll
  for (int j = 0; j < 4; ++j) {
    const int n = n0 + ty + 8 * j, k = k0 + tx;
    if (k < K && n < N) Wt[(size_t)n * K + k] = tile[tx][ty + 8 * j];
  }
}

// ---------------------------------------------------------------------------
// QKV GEMM: [8192 x 768] x [768 x 2304] + bias; 64x64 tile per wave.
// Scatters Q,K as [B,H,S,HS] (f16) and V transposed as [B,H,HS,S] (f16).
__global__ __launch_bounds__(256)
void qkv_gemm(const _Float16* __restrict__ xh, const _Float16* __restrict__ wT,
              const float* __restrict__ bias, _Float16* __restrict__ Qh,
              _Float16* __restrict__ Kh, _Float16* __restrict__ Vt) {
  const int wid = blockIdx.x * 8 + (threadIdx.x >> 5);
  const int NT = N3 / 64;  // 36 n-tiles
  const int mt = wid / NT, nt = wid % NT;
  if (mt >= Mrows / 64) return;
  const int lane = threadIdx.x & 31, half = lane >> 4, lm = lane & 15;
  const int m0 = mt * 64, n0 = nt * 64;

  v8f acc[4][4] = {};
  for (int k0 = 0; k0 < Dq; k0 += 32) {
    v16h a[4], b[4];
#pragma unroll
    for (int mi = 0; mi < 4; ++mi) {
      const _Float16* p = xh + (size_t)(m0 + 16 * mi + lm) * Dq + k0 + 8 * half;
      v8h lo = *(const v8h*)p;
      v8h hi = *(const v8h*)(p + 16);
#pragma unroll
      for (int e = 0; e < 8; ++e) { a[mi][e] = lo[e]; a[mi][8 + e] = hi[e]; }
    }
#pragma unroll
    for (int ni = 0; ni < 4; ++ni) {
      const _Float16* p = wT + (size_t)(n0 + 16 * ni + lm) * Dq + k0 + 8 * half;
      v8h lo = *(const v8h*)p;
      v8h hi = *(const v8h*)(p + 16);
#pragma unroll
      for (int e = 0; e < 8; ++e) { b[ni][e] = lo[e]; b[ni][8 + e] = hi[e]; }
    }
#pragma unroll
    for (int mi = 0; mi < 4; ++mi)
#pragma unroll
      for (int ni = 0; ni < 4; ++ni)
        acc[mi][ni] = WMMA_F16(a[mi], b[ni], acc[mi][ni]);
  }

#pragma unroll
  for (int mi = 0; mi < 4; ++mi) {
#pragma unroll
    for (int ni = 0; ni < 4; ++ni) {
#pragma unroll
      for (int e = 0; e < 8; ++e) {
        const int r = m0 + 16 * mi + 8 * half + e;   // C layout: M = e + 8*half
        const int c = n0 + 16 * ni + lm;             // N = lane&15
        const float v = acc[mi][ni][e] + bias[c];
        const int bb = r >> 10, s = r & 1023;
        const int t = c / Dq, rem = c % Dq, h = rem >> 6, hs = rem & 63;
        const _Float16 hv = (_Float16)v;
        if (t == 0)      Qh[((size_t)(bb * Hq + h) * Sq + s) * HSq + hs] = hv;
        else if (t == 1) Kh[((size_t)(bb * Hq + h) * Sq + s) * HSq + hs] = hv;
        else             Vt[((size_t)(bb * Hq + h) * HSq + hs) * Sq + s] = hv;
      }
    }
  }
}

// ---------------------------------------------------------------------------
// Flash attention: one wave per (b, h, 32-query tile); K/V fragments are
// reused across two query M-tiles. Single-wave workgroup => __syncthreads()
// is hardware-NOP but fences the LDS transpose.
__global__ __launch_bounds__(32)
void attn(const _Float16* __restrict__ Qh, const _Float16* __restrict__ Kh,
          const _Float16* __restrict__ Vt, _Float16* __restrict__ Oh) {
  __shared__ _Float16 pl[2 * 32 * 16];  // per M-tile P, column-major [key][row]
  const int lane = threadIdx.x & 31, half = lane >> 4, lm = lane & 15;
  const int qt = blockIdx.x & 31;           // S/32 = 32 query tiles
  const int bh = blockIdx.x >> 5;           // 0..95
  const int bb = bh / Hq, h = bh % Hq;
  const _Float16* Qp = Qh + (size_t)bh * Sq * HSq;
  const _Float16* Kp = Kh + (size_t)bh * Sq * HSq;
  const _Float16* Vp = Vt + (size_t)bh * HSq * Sq;
  const int q0 = qt * 32;

  // Q tile (32 x 64) as 2 M-tiles x 2 K-steps of A-fragments, loaded once
  v16h qa[2][2];
#pragma unroll
  for (int mi = 0; mi < 2; ++mi)
#pragma unroll
    for (int ks = 0; ks < 2; ++ks) {
      const _Float16* p = Qp + (size_t)(q0 + 16 * mi + lm) * HSq + 32 * ks + 8 * half;
      v8h lo = *(const v8h*)p;
      v8h hi = *(const v8h*)(p + 16);
#pragma unroll
      for (int e = 0; e < 8; ++e) { qa[mi][ks][e] = lo[e]; qa[mi][ks][8 + e] = hi[e]; }
    }

  float mrow[2][8], lrow[2][8];
#pragma unroll
  for (int mi = 0; mi < 2; ++mi)
#pragma unroll
    for (int e = 0; e < 8; ++e) { mrow[mi][e] = -1e30f; lrow[mi][e] = 0.f; }
  v8f o[2][4] = {};

  const int nkb = qt + 1;  // causal: key blocks 0..qt (kt0 = q0 on the diagonal)
  for (int kb = 0; kb < nkb; ++kb) {
    const int kt0 = kb * 32;
    // scores: 32 queries x 32 keys (2x2 C tiles); K fragments shared over mi
    v8f sc[2][2] = {};
#pragma unroll
    for (int ks = 0; ks < 2; ++ks) {
#pragma unroll
      for (int ni = 0; ni < 2; ++ni) {
        const _Float16* p = Kp + (size_t)(kt0 + 16 * ni + lm) * HSq + 32 * ks + 8 * half;
        v8h lo = *(const v8h*)p;
        v8h hi = *(const v8h*)(p + 16);
        v16h bf;
#pragma unroll
        for (int e = 0; e < 8; ++e) { bf[e] = lo[e]; bf[8 + e] = hi[e]; }
#pragma unroll
        for (int mi = 0; mi < 2; ++mi)
          sc[mi][ni] = WMMA_F16(qa[mi][ks], bf, sc[mi][ni]);
      }
    }
    // scale; only the diagonal block needs the causal mask (wave-uniform branch)
    if (kb == qt) {
#pragma unroll
      for (int mi = 0; mi < 2; ++mi)
#pragma unroll
        for (int ni = 0; ni < 2; ++ni)
#pragma unroll
          for (int e = 0; e < 8; ++e) {
            const int qi = q0 + 16 * mi + 8 * half + e, ki = kt0 + 16 * ni + lm;
            sc[mi][ni][e] = (ki <= qi) ? sc[mi][ni][e] * 0.125f : -1e30f;
          }
    } else {
#pragma unroll
      for (int mi = 0; mi < 2; ++mi)
#pragma unroll
        for (int ni = 0; ni < 2; ++ni)
#pragma unroll
          for (int e = 0; e < 8; ++e) sc[mi][ni][e] *= 0.125f;
    }
    // online softmax per M-tile
#pragma unroll
    for (int mi = 0; mi < 2; ++mi) {
      float mnew[8];
#pragma unroll
      for (int e = 0; e < 8; ++e)
        mnew[e] = fmaxf(mrow[mi][e], fmaxf(sc[mi][0][e], sc[mi][1][e]));
#pragma unroll
      for (int msk = 1; msk < 16; msk <<= 1)
#pragma unroll
        for (int e = 0; e < 8; ++e) mnew[e] = fmaxf(mnew[e], __shfl_xor(mnew[e], msk, 32));
      float alpha[8], rsum[8];
#pragma unroll
      for (int e = 0; e < 8; ++e) { alpha[e] = __expf(mrow[mi][e] - mnew[e]); rsum[e] = 0.f; }
#pragma unroll
      for (int ni = 0; ni < 2; ++ni)
#pragma unroll
        for (int e = 0; e < 8; ++e) {
          const float pv = __expf(sc[mi][ni][e] - mnew[e]);
          sc[mi][ni][e] = pv;
          rsum[e] += pv;
        }
#pragma unroll
      for (int msk = 1; msk < 16; msk <<= 1)
#pragma unroll
        for (int e = 0; e < 8; ++e) rsum[e] += __shfl_xor(rsum[e], msk, 32);
#pragma unroll
      for (int e = 0; e < 8; ++e) {
        lrow[mi][e] = lrow[mi][e] * alpha[e] + rsum[e];
        mrow[mi][e] = mnew[e];
      }
#pragma unroll
      for (int nt = 0; nt < 4; ++nt)
#pragma unroll
        for (int e = 0; e < 8; ++e) o[mi][nt][e] *= alpha[e];

      // P: C layout -> LDS (column-major per M-tile; 2 x b128 per lane)
#pragma unroll
      for (int ni = 0; ni < 2; ++ni) {
        v8h ph;
#pragma unroll
        for (int e = 0; e < 8; ++e) ph[e] = (_Float16)sc[mi][ni][e];
        *(v8h*)(pl + mi * 512 + (16 * ni + lm) * 16 + 8 * half) = ph;
      }
    }
    __syncthreads();
    v16h pa[2];
#pragma unroll
    for (int mi = 0; mi < 2; ++mi)
#pragma unroll
      for (int i = 0; i < 16; ++i) {
        const int kidx = 16 * (i >> 3) + 8 * half + 2 * ((i >> 1) & 3) + (i & 1);
        pa[mi][i] = pl[mi * 512 + kidx * 16 + lm];
      }
    __syncthreads();
    // O += P(32x32) x V(32x64): V fragments shared over both M-tiles
#pragma unroll
    for (int nt = 0; nt < 4; ++nt) {
      const _Float16* p = Vp + (size_t)(16 * nt + lm) * Sq + kt0 + 8 * half;
      v8h lo = *(const v8h*)p;
      v8h hi = *(const v8h*)(p + 16);
      v16h vb;
#pragma unroll
      for (int e = 0; e < 8; ++e) { vb[e] = lo[e]; vb[8 + e] = hi[e]; }
#pragma unroll
      for (int mi = 0; mi < 2; ++mi)
        o[mi][nt] = WMMA_F16(pa[mi], vb, o[mi][nt]);
    }
  }

  // epilogue: O row-major [B,S,D] as f16 for the projection GEMM
#pragma unroll
  for (int mi = 0; mi < 2; ++mi)
#pragma unroll
    for (int nt = 0; nt < 4; ++nt)
#pragma unroll
      for (int e = 0; e < 8; ++e) {
        const int s = q0 + 16 * mi + 8 * half + e;
        const float v = o[mi][nt][e] / lrow[mi][e];
        Oh[((size_t)(bb * Sq + s)) * Dq + h * HSq + 16 * nt + lm] = (_Float16)v;
      }
}

// ---------------------------------------------------------------------------
// Output projection: [8192 x 768] x [768 x 768] + bias -> fp32 out
__global__ __launch_bounds__(256)
void proj_gemm(const _Float16* __restrict__ Oh, const _Float16* __restrict__ wT,
               const float* __restrict__ bias, float* __restrict__ out) {
  const int wid = blockIdx.x * 8 + (threadIdx.x >> 5);
  const int NT = Dq / 64;  // 12 n-tiles
  const int mt = wid / NT, nt = wid % NT;
  if (mt >= Mrows / 64) return;
  const int lane = threadIdx.x & 31, half = lane >> 4, lm = lane & 15;
  const int m0 = mt * 64, n0 = nt * 64;

  v8f acc[4][4] = {};
  for (int k0 = 0; k0 < Dq; k0 += 32) {
    v16h a[4], b[4];
#pragma unroll
    for (int mi = 0; mi < 4; ++mi) {
      const _Float16* p = Oh + (size_t)(m0 + 16 * mi + lm) * Dq + k0 + 8 * half;
      v8h lo = *(const v8h*)p;
      v8h hi = *(const v8h*)(p + 16);
#pragma unroll
      for (int e = 0; e < 8; ++e) { a[mi][e] = lo[e]; a[mi][8 + e] = hi[e]; }
    }
#pragma unroll
    for (int ni = 0; ni < 4; ++ni) {
      const _Float16* p = wT + (size_t)(n0 + 16 * ni + lm) * Dq + k0 + 8 * half;
      v8h lo = *(const v8h*)p;
      v8h hi = *(const v8h*)(p + 16);
#pragma unroll
      for (int e = 0; e < 8; ++e) { b[ni][e] = lo[e]; b[ni][8 + e] = hi[e]; }
    }
#pragma unroll
    for (int mi = 0; mi < 4; ++mi)
#pragma unroll
      for (int ni = 0; ni < 4; ++ni)
        acc[mi][ni] = WMMA_F16(a[mi], b[ni], acc[mi][ni]);
  }

#pragma unroll
  for (int mi = 0; mi < 4; ++mi)
#pragma unroll
    for (int ni = 0; ni < 4; ++ni)
#pragma unroll
      for (int e = 0; e < 8; ++e) {
        const int r = m0 + 16 * mi + 8 * half + e;
        const int c = n0 + 16 * ni + lm;
        out[(size_t)r * Dq + c] = acc[mi][ni][e] + bias[c];
      }
}

// ---------------------------------------------------------------------------
extern "C" void kernel_launch(void* const* d_in, const int* in_sizes, int n_in,
                              void* d_out, int out_size, void* d_ws, size_t ws_size,
                              hipStream_t stream) {
  (void)in_sizes; (void)n_in; (void)out_size; (void)ws_size;
  const float* x      = (const float*)d_in[0];
  const float* W_attn = (const float*)d_in[1];
  const float* b_attn = (const float*)d_in[2];
  const float* W_proj = (const float*)d_in[3];
  const float* b_proj = (const float*)d_in[4];
  float* out = (float*)d_out;

  // Workspace layout (f16 elements), all 16B-aligned
  _Float16* ws  = (_Float16*)d_ws;
  _Float16* xh  = ws;                              // 8192*768
  _Float16* WaT = xh  + (size_t)Mrows * Dq;        // 2304*768 (transposed)
  _Float16* WpT = WaT + (size_t)N3 * Dq;           // 768*768  (transposed)
  _Float16* Qh  = WpT + (size_t)Dq * Dq;           // [B,H,S,HS]
  _Float16* Kh  = Qh  + (size_t)Mrows * Dq;        // [B,H,S,HS]
  _Float16* Vt  = Kh  + (size_t)Mrows * Dq;        // [B,H,HS,S]
  _Float16* Oh  = Vt  + (size_t)Mrows * Dq;        // [B,S,D]

  const int nx = Mrows * Dq;
  cvt_f32_f16<<<(nx + 255) / 256, 256, 0, stream>>>(x, xh, nx);
  {
    dim3 blk(32, 8);
    dim3 grdA(Dq / 32, N3 / 32);  // 24 x 72
    dim3 grdP(Dq / 32, Dq / 32);  // 24 x 24
    cvt_transpose<<<grdA, blk, 0, stream>>>(W_attn, WaT, Dq, N3);
    cvt_transpose<<<grdP, blk, 0, stream>>>(W_proj, WpT, Dq, Dq);
  }

  // 128 m-tiles x 36 n-tiles, 8 waves/block
  qkv_gemm<<<(128 * 36) / 8, 256, 0, stream>>>(xh, WaT, b_attn, Qh, Kh, Vt);

  // one wave per (b, h, 32-query tile): 8*12*32 = 3072 blocks
  attn<<<Bq * Hq * (Sq / 32), 32, 0, stream>>>(Qh, Kh, Vt, Oh);

  // 128 m-tiles x 12 n-tiles, 8 waves/block
  proj_gemm<<<(128 * 12) / 8, 256, 0, stream>>>(Oh, WpT, b_proj, out);
}
